// Attention_67637144977803
// MI455X (gfx1250) — compile-verified, matrix operand order checked by reading, not run on hardware
//
#include <hip/hip_runtime.h>
#include <hip/hip_bf16.h>

typedef _Float16 f16;
typedef __attribute__((ext_vector_type(8)))  _Float16 v8h;
typedef __attribute__((ext_vector_type(16))) _Float16 v16h;
typedef __attribute__((ext_vector_type(8)))  float    v8f;

#define NHEADS   8
#define HDIM     64
#define HIDDEN   512
#define CDIM     256
#define NTOK     2304   // 48*48
#define BATCH    2

// ---------------------------------------------------------------------------
// Fragment loader: CDNA5 16-bit A/B WMMA fragment (16x32 or 32x16) from LDS.
// Per lane: row/col = lane&15 ; K elements {kk..kk+7} u {kk+16..kk+23},
// kk = (lane<16)?0:8  -> two 16B (ds_load_b128) reads.
// stride must be a multiple of 8 f16 (16B alignment).
// ---------------------------------------------------------------------------
__device__ __forceinline__ v16h load_frag(const f16* base, int lane, int stride, int ks) {
    const int row = lane & 15;
    const int kk  = (lane < 16) ? 0 : 8;
    const f16* p  = base + row * stride + ks + kk;
    v8h lo = *reinterpret_cast<const v8h*>(p);
    v8h hi = *reinterpret_cast<const v8h*>(p + 16);
    return __builtin_shufflevector(lo, hi, 0,1,2,3,4,5,6,7,8,9,10,11,12,13,14,15);
}

__device__ __forceinline__ v8f wmma16(v16h a, v16h b, v8f c) {
    return __builtin_amdgcn_wmma_f32_16x16x32_f16(false, a, false, b, (short)0, c, false, false);
}

// Async global->LDS 16B copy (ASYNCcnt path). GVS mode: uniform 64-bit SGPR
// base + per-lane 32-bit byte offset; VDST carries the LDS byte address
// (low 32 bits of the generic shared pointer).
__device__ __forceinline__ void async_b128_to_lds(const f16* lds_dst, const f16* gbase,
                                                  unsigned byte_off) {
    unsigned lds_addr = (unsigned)(unsigned long long)lds_dst;
    asm volatile("global_load_async_to_lds_b128 %0, %1, %2 offset:0"
                 :: "v"(lds_addr), "v"(byte_off), "s"(gbase)
                 : "memory");
}
__device__ __forceinline__ void wait_async0() {
    asm volatile("s_wait_asynccnt 0" ::: "memory");
}

// ---------------------------------------------------------------------------
// fp32 -> f16 elementwise convert
// ---------------------------------------------------------------------------
__global__ void cvt_f32_f16(const float* __restrict__ src, f16* __restrict__ dst, int n) {
    int i = blockIdx.x * blockDim.x + threadIdx.x;
    if (i < n) dst[i] = (f16)src[i];
}

// ---------------------------------------------------------------------------
// Generic WMMA GEMM: C[M,N] = A[M,K] * B[K,N] (+bias[M]); f16 in, f32 out.
// Batched over blockIdx.z (A shared, B/C strided). 256 thr = 8 waves.
// Block tile 128x128, k-step 32; each wave: 32(M) x 64(N) = 8 accumulators.
// A-tile copied with global_load_async_to_lds_b128 (non-transposing);
// B-tile transposed element-wise through VGPRs.
// Requires M%128==0, N%128==0, K%32==0 (true for all uses here).
// ---------------------------------------------------------------------------
__global__ __launch_bounds__(256)
void gemm_f16_wmma(const f16* __restrict__ A, const f16* __restrict__ B,
                   float* __restrict__ C, const float* __restrict__ bias,
                   int M, int N, int K, long strideB, long strideC) {
    __shared__ f16 at[128 * 40];   // [m][k] k-step 32, stride 40 (80B, 16B aligned)
    __shared__ f16 bt[128 * 40];   // [n][k] transposed

    const int tid  = threadIdx.x;
    const int lane = tid & 31;
    const int wave = tid >> 5;
    const int wr   = wave >> 1;            // 0..3 -> M offset *32
    const int wc   = wave & 1;             // 0..1 -> N offset *64
    const int m0   = blockIdx.y * 128;
    const int n0   = blockIdx.x * 128;
    const f16* Bb  = B + (long)blockIdx.z * strideB;
    float*     Cb  = C + (long)blockIdx.z * strideC;

    v8f acc[2][4] = {};

    for (int ks = 0; ks < K; ks += 32) {
        __syncthreads();
        // A tile 128x32: 512 x 16B async chunks, 2 per thread
#pragma unroll
        for (int i = 0; i < 2; i++) {
            int idx = tid + i * 256;
            int r = idx >> 2, kc = (idx & 3) * 8;
            async_b128_to_lds(&at[r * 40 + kc], A,
                              (unsigned)(((m0 + r) * K + ks + kc) * 2));
        }
        // B tile 32x128 -> [n][k] (transpose; coalesced over n)
        for (int idx = tid; idx < 32 * 128; idx += 256) {
            int nc = idx & 127, kr = idx >> 7;
            bt[nc * 40 + kr] = Bb[(long)(ks + kr) * N + n0 + nc];
        }
        // prefetch next k-step (global_prefetch_b8)
        if (ks + 32 < K) {
            __builtin_prefetch(&A[(long)(m0 + (tid >> 1)) * K + ks + 32], 0, 1);
            __builtin_prefetch(&Bb[(long)(ks + 32 + (tid >> 5)) * N + n0 + (lane << 2)], 0, 1);
        }
        wait_async0();
        __syncthreads();

        v16h a0 = load_frag(at + (wr * 32)      * 40, lane, 40, 0);
        v16h a1 = load_frag(at + (wr * 32 + 16) * 40, lane, 40, 0);
#pragma unroll
        for (int j = 0; j < 4; j++) {
            v16h bf = load_frag(bt + (wc * 64 + j * 16) * 40, lane, 40, 0);
            acc[0][j] = wmma16(a0, bf, acc[0][j]);
            acc[1][j] = wmma16(a1, bf, acc[1][j]);
        }
    }

    const int half = lane >> 4, ln = lane & 15;
#pragma unroll
    for (int i = 0; i < 2; i++)
#pragma unroll
        for (int j = 0; j < 4; j++)
#pragma unroll
            for (int r = 0; r < 8; r++) {
                int row = m0 + wr * 32 + i * 16 + r + 8 * half;
                int col = n0 + wc * 64 + j * 16 + ln;
                float v = acc[i][j][r];
                if (bias) v += bias[row];
                Cb[(long)row * N + col] = v;
            }
}

// ---------------------------------------------------------------------------
// Per-row l2 norm over tokens for q,k rows (rows 0..1023 of each batch's 1536),
// plus fp32->f16 quantization of all of qkv. One block per row.
// ---------------------------------------------------------------------------
__global__ __launch_bounds__(256)
void norm_quant_qkv(const float* __restrict__ qkv, f16* __restrict__ qkvh) {
    const int row = blockIdx.x;            // b*1536 + r
    const int r   = row % 1536;
    const float* src = qkv  + (long)row * NTOK;
    f16*         dst = qkvh + (long)row * NTOK;

    if (r < 1024) {                        // q or k row: normalize over tokens
        __shared__ float red[256];
        float ss = 0.f;
        for (int i = threadIdx.x; i < NTOK; i += 256) { float v = src[i]; ss += v * v; }
        red[threadIdx.x] = ss;
        __syncthreads();
        for (int s = 128; s > 0; s >>= 1) {
            if (threadIdx.x < s) red[threadIdx.x] += red[threadIdx.x + s];
            __syncthreads();
        }
        float inv = 1.0f / fmaxf(sqrtf(red[0]), 1e-12f);
        for (int i = threadIdx.x; i < NTOK; i += 256) dst[i] = (f16)(src[i] * inv);
    } else {                               // v row: convert only
        for (int i = threadIdx.x; i < NTOK; i += 256) dst[i] = (f16)src[i];
    }
}

// ---------------------------------------------------------------------------
// Fused flash attention per (b, head). q,k,v stored (d, n) d-major, f16.
// Block = 128 thr (4 waves); each wave owns 16 query tokens; block covers 64.
// Key blocks of 32 tokens; online softmax; O (16x64) in 4 f32 accumulators.
// V-tile copied via global_load_async_to_lds_b128 (non-transposing).
// ---------------------------------------------------------------------------
__global__ __launch_bounds__(128)
void flash_attn(const f16* __restrict__ qkvh, f16* __restrict__ outh) {
    __shared__ f16 qtile[64 * 72];      // [token][dim], stride 72 (144B, 16B aligned)
    __shared__ f16 ktile[32 * 72];      // [token][dim]
    __shared__ f16 vtile[64 * 40];      // [dim][token], stride 40
    __shared__ f16 ptile[4][16 * 40];   // per-wave P tile [qrow][ktok]

    const int tid  = threadIdx.x;
    const int lane = tid & 31;
    const int wave = tid >> 5;
    const int n0   = blockIdx.x * 64;
    const int h    = blockIdx.y;
    const int b    = blockIdx.z;

    const long headoff = ((long)b * 1536 + h * HDIM) * NTOK;
    const f16* q = qkvh + headoff;
    const f16* k = qkvh + headoff + (long)512  * NTOK;
    const f16* v = qkvh + headoff + (long)1024 * NTOK;

    // load Q tile (64 tokens x 64 dims), transpose (d,n)->(n,d)
    for (int idx = tid; idx < 64 * 64; idx += 128) {
        int tl = idx & 63, d = idx >> 6;
        qtile[tl * 72 + d] = q[(long)d * NTOK + n0 + tl];
    }

    const int wq0 = wave * 16;
    float mrow[8], lrow[8];
#pragma unroll
    for (int r = 0; r < 8; r++) { mrow[r] = -3.0e38f; lrow[r] = 0.f; }
    v8f o[4] = {};

    f16* pt = &ptile[wave][0];
    const int half = lane >> 4, ln = lane & 15;

    for (int kb = 0; kb < NTOK; kb += 32) {
        __syncthreads();
        // V tile [dim][token]: 256 x 16B async chunks, 2 per thread
#pragma unroll
        for (int i = 0; i < 2; i++) {
            int idx = tid + i * 128;
            int d = idx >> 2, c = (idx & 3) * 8;
            async_b128_to_lds(&vtile[d * 40 + c], v,
                              (unsigned)((d * NTOK + kb + c) * 2));
        }
        // K tile [token][dim] (transpose)
        for (int idx = tid; idx < 32 * 64; idx += 128) {
            int tl = idx & 31, d = idx >> 5;
            ktile[tl * 72 + d] = k[(long)d * NTOK + kb + tl];
        }
        if (kb + 32 < NTOK) {
            __builtin_prefetch(&k[(long)(tid >> 1) * NTOK + kb + 32 + (tid & 1) * 16], 0, 1);
            __builtin_prefetch(&v[(long)(tid >> 1) * NTOK + kb + 32 + (tid & 1) * 16], 0, 1);
        }
        wait_async0();
        __syncthreads();

        // S = Q'(16x64) x K'^T(64x32) : two col tiles, two k-steps each
        v8f s0 = {}, s1 = {};
#pragma unroll
        for (int ks = 0; ks < HDIM; ks += 32) {
            v16h aq = load_frag(qtile + wq0 * 72, lane, 72, ks);
            v16h b0 = load_frag(ktile,           lane, 72, ks);
            v16h b1 = load_frag(ktile + 16 * 72, lane, 72, ks);
            s0 = wmma16(aq, b0, s0);
            s1 = wmma16(aq, b1, s1);
        }

        // online softmax; VGPR r <-> query row (r + 8*half); lanes = columns
#pragma unroll
        for (int r = 0; r < 8; r++) {
            float x0 = s0[r], x1 = s1[r];
            float mx = fmaxf(x0, x1);
#pragma unroll
            for (int off = 8; off > 0; off >>= 1) mx = fmaxf(mx, __shfl_xor(mx, off, 32));
            float mnew = fmaxf(mrow[r], mx);
            float p0 = __expf(x0 - mnew);
            float p1 = __expf(x1 - mnew);
            float rs = p0 + p1;
#pragma unroll
            for (int off = 8; off > 0; off >>= 1) rs += __shfl_xor(rs, off, 32);
            float alpha = __expf(mrow[r] - mnew);
            mrow[r] = mnew;
            lrow[r] = lrow[r] * alpha + rs;
            o[0][r] *= alpha; o[1][r] *= alpha; o[2][r] *= alpha; o[3][r] *= alpha;
            int rr = r + 8 * half;
            pt[rr * 40 + ln]      = (f16)p0;
            pt[rr * 40 + ln + 16] = (f16)p1;
        }
        // wave-local LDS RAW (cross-lane): wait for our ds stores to land
        asm volatile("s_wait_dscnt 0" ::: "memory");

        // O += P(16x32) x V'^T(32x64): one K=32 WMMA per 16-dim tile
        v16h pa = load_frag(pt, lane, 40, 0);
#pragma unroll
        for (int j = 0; j < 4; j++) {
            v16h vb = load_frag(vtile + (j * 16) * 40, lane, 40, 0);
            o[j] = wmma16(pa, vb, o[j]);
        }
    }

    // epilogue: divide by row sums, store as (d, n) f16 for projection GEMM
    const long outbase = ((long)b * HIDDEN + h * HDIM) * NTOK;
#pragma unroll
    for (int j = 0; j < 4; j++)
#pragma unroll
        for (int r = 0; r < 8; r++) {
            float val = o[j][r] / lrow[r];
            int d = j * 16 + ln;
            int n = n0 + wq0 + r + 8 * half;
            outh[outbase + (long)d * NTOK + n] = (f16)val;
        }
}

// ---------------------------------------------------------------------------
extern "C" void kernel_launch(void* const* d_in, const int* in_sizes, int n_in,
                              void* d_out, int out_size, void* d_ws, size_t ws_size,
                              hipStream_t stream) {
    const float* x      = (const float*)d_in[0];   // (2,256,48,48)
    const float* w_qkv  = (const float*)d_in[1];   // (1536,256)
    const float* w_proj = (const float*)d_in[2];   // (256,512)
    const float* b_proj = (const float*)d_in[3];   // (256,)
    float* out = (float*)d_out;                    // (2,256,48,48)

    const int nx     = BATCH * CDIM * NTOK;        // 1179648
    const int nwqkv  = 1536 * CDIM;                // 393216
    const int nwproj = CDIM * HIDDEN;              // 131072
    const long nqkv  = (long)BATCH * 1536 * NTOK;  // 7077888
    const long nouth = (long)BATCH * HIDDEN * NTOK;

    char* ws = (char*)d_ws;
    size_t off = 0;
    auto alloc = [&](size_t bytes) { void* p = ws + off; off = (off + bytes + 255) & ~(size_t)255; return p; };

    f16*   xh     = (f16*)  alloc((size_t)nx * 2);
    f16*   wqkvh  = (f16*)  alloc((size_t)nwqkv * 2);
    f16*   wprojh = (f16*)  alloc((size_t)nwproj * 2);
    float* qkv32  = (float*)alloc((size_t)nqkv * 4);
    f16*   qkvh   = (f16*)  alloc((size_t)nqkv * 2);
    f16*   outh   = (f16*)  alloc((size_t)nouth * 2);

    // 1) fp32 -> f16 conversions
    cvt_f32_f16<<<(nx     + 255) / 256, 256, 0, stream>>>(x,      xh,     nx);
    cvt_f32_f16<<<(nwqkv  + 255) / 256, 256, 0, stream>>>(w_qkv,  wqkvh,  nwqkv);
    cvt_f32_f16<<<(nwproj + 255) / 256, 256, 0, stream>>>(w_proj, wprojh, nwproj);

    // 2) QKV GEMM: per batch, (1536x256) x (256x2304) -> fp32
    gemm_f16_wmma<<<dim3(NTOK / 128, 1536 / 128, BATCH), 256, 0, stream>>>(
        wqkvh, xh, qkv32, nullptr, 1536, NTOK, CDIM,
        (long)CDIM * NTOK, (long)1536 * NTOK);

    // 3) l2-norm q,k over tokens + quantize qkv to f16
    norm_quant_qkv<<<BATCH * 1536, 256, 0, stream>>>(qkv32, qkvh);

    // 4) fused flash attention per (b, head)
    flash_attn<<<dim3(NTOK / 64, NHEADS, BATCH), 128, 0, stream>>>(qkvh, outh);

    // 5) projection: (256x512) x (512x2304) + bias -> d_out (fp32)
    gemm_f16_wmma<<<dim3(NTOK / 128, CDIM / 128, BATCH), 256, 0, stream>>>(
        wprojh, outh, out, b_proj, CDIM, NTOK, HIDDEN,
        (long)HIDDEN * NTOK, (long)CDIM * NTOK);
}
